// AttentionModule_73409581023565
// MI455X (gfx1250) — matrix operand order repaired and checked by reading, not compiled, not run on hardware
//
#include <hip/hip_runtime.h>
#include <hip/hip_bf16.h>
#include <math.h>

// Problem constants (fixed by the reference)
#define Bd  4
#define Sd  2048
#define Dd  1024
#define Hd  16
#define HDd 64
#define Md  (Bd*Sd)   // 8192 rows

typedef __attribute__((ext_vector_type(16))) __bf16 v16bf;
typedef __attribute__((ext_vector_type(8)))  float  v8f;
typedef __attribute__((ext_vector_type(4)))  unsigned int u32x4;
typedef __attribute__((ext_vector_type(8)))  int   i32x8;
typedef __attribute__((ext_vector_type(4)))  int   i32x4;

union AB16 { v16bf v; uint4 q[2]; };

static __device__ __forceinline__ unsigned short f2bf(float f) {
  unsigned int u = __float_as_uint(f);
  u += 0x7FFFu + ((u >> 16) & 1u);   // round-to-nearest-even
  return (unsigned short)(u >> 16);
}

// ---------------------------------------------------------------------------
// TDM: DMA a 64-row x 32-col bf16 tile (row stride K=1024 elems) into LDS.
// D# per CDNA5 ISA 8.3/8.4: group0 = count/lds_addr/global_addr/type,
// group1 = data_size, tensor dims, tile dims, strides. 2-D tile -> groups 2,3 zero.
// This toolchain exposes the 6-arg builtin (extra i32x8 group before cpol).
// ---------------------------------------------------------------------------
static __device__ __forceinline__ void tdm_load_a_tile(const unsigned short* gsrc,
                                                       unsigned lds_byte_off) {
  unsigned long long ga = (unsigned long long)(uintptr_t)gsrc;
  u32x4 g0;
  g0[0] = 1u;                                             // count=1, is_restore=0
  g0[1] = lds_byte_off;                                   // lds_addr (bytes)
  g0[2] = (unsigned)(ga & 0xFFFFFFFFu);                   // global_addr[31:0]
  g0[3] = (unsigned)((ga >> 32) & 0x1FFFFFFu) | (2u << 30); // global_addr[56:32] | type=2
  i32x8 g1;
  g1[0] = 1 << 16;                                        // data_size=1 (2 bytes)
  g1[1] = (int)((1024u & 0xFFFFu) << 16);                 // tensor_dim0 lo16 (=K)
  g1[2] = (int)(((1024u >> 16) & 0xFFFFu) | ((8192u & 0xFFFFu) << 16)); // dim0 hi | dim1 lo (=M)
  g1[3] = (int)(((8192u >> 16) & 0xFFFFu) | (32u << 16)); // dim1 hi | tile_dim0=32
  g1[4] = 64;                                             // tile_dim1=64, tile_dim2=0
  g1[5] = 1024;                                           // tensor_dim0_stride lo32 (=K)
  g1[6] = 0;                                              // stride0 hi | stride1 lo
  g1[7] = 0;                                              // stride1 hi
  i32x4 z4 = {0, 0, 0, 0};
  i32x8 z8 = {0, 0, 0, 0, 0, 0, 0, 0};
  __builtin_amdgcn_tensor_load_to_lds(g0, g1, z4, z4, z8, 0);
}

// ---------------------------------------------------------------------------
// xp = bf16(x + pe)   (8M elements, 4 per thread)
// ---------------------------------------------------------------------------
__global__ void pe_add_bf16(const float* __restrict__ x, const float* __restrict__ pe,
                            unsigned short* __restrict__ xp) {
  long long i4 = (long long)blockIdx.x * blockDim.x + threadIdx.x;
  long long e  = i4 * 4;
  int  d = (int)(e & (Dd - 1));
  long long m = e >> 10;              // row in [0, 8192)
  int  s = (int)(m & (Sd - 1));
  float4 xv = *(const float4*)(x + e);
  float4 pv = *(const float4*)(pe + (long long)s * Dd + d);
  ushort4 o;
  o.x = f2bf(xv.x + pv.x); o.y = f2bf(xv.y + pv.y);
  o.z = f2bf(xv.z + pv.z); o.w = f2bf(xv.w + pv.w);
  *(ushort4*)(xp + e) = o;
}

// fp32 -> bf16 (weights), 4 per thread
__global__ void cvt_bf16(const float* __restrict__ src, unsigned short* __restrict__ dst) {
  long long i4 = (long long)blockIdx.x * blockDim.x + threadIdx.x;
  long long e  = i4 * 4;
  float4 v = *(const float4*)(src + e);
  ushort4 o;
  o.x = f2bf(v.x); o.y = f2bf(v.y); o.z = f2bf(v.z); o.w = f2bf(v.w);
  *(ushort4*)(dst + e) = o;
}

// ---------------------------------------------------------------------------
// GEMM: out[m,n] = sum_k A[m,k] * W[n,k] + bias[n]
//   A [8192,1024] bf16 row-major, W [1024,1024] bf16 row-major.
// Block = 8 waves sharing one 64-row A slab; the slab is staged into LDS by
// the Tensor Data Mover (double-buffered), B comes straight from L2.
// Each wave owns a 64x64 output tile (4x4 WMMA accumulators), K-step 32.
// ---------------------------------------------------------------------------
#define OUT_QK  0
#define OUT_VT  1
#define OUT_F32 2

__global__ void __launch_bounds__(256)
gemm_bf16_wmma(const unsigned short* __restrict__ A,
               const unsigned short* __restrict__ W,
               const float* __restrict__ bias,
               void* __restrict__ out, int mode) {
  __shared__ unsigned short ldsA[2][64 * 32];   // two 4KB A slabs (double buffer)
  const int K = Dd, N = Dd;
  const int lane  = threadIdx.x & 31;
  const int half  = lane >> 4;        // 0: lanes 0-15, 1: lanes 16-31
  const int lrow  = lane & 15;
  const int wslot = threadIdx.x >> 5;
  const int wg    = blockIdx.x * (blockDim.x >> 5) + wslot;
  const int tilesN = N / 64;          // 16
  const int m0 = (wg / tilesN) * 64;  // same for all 8 waves in the block
  const int n0 = (wg % tilesN) * 64;

  const unsigned lds0 = (unsigned)(uintptr_t)&ldsA[0][0];
  const unsigned lds1 = (unsigned)(uintptr_t)&ldsA[1][0];

  const v8f vzero = {0.f,0.f,0.f,0.f,0.f,0.f,0.f,0.f};
  v8f acc[4][4];
#pragma unroll
  for (int i = 0; i < 4; ++i)
#pragma unroll
    for (int j = 0; j < 4; ++j) acc[i][j] = vzero;

  const unsigned short* Aslab = A + (long long)m0 * K;   // block's 64-row slab
  const unsigned short* Wbase = W + (long long)n0 * K;

  // prologue: stage k0=0 slab
  if (wslot == 0) {
    tdm_load_a_tile(Aslab, lds0);
    __builtin_amdgcn_s_wait_tensorcnt(0);
  }
  __syncthreads();

  for (int k0 = 0; k0 < K; k0 += 32) {
    const int cur = (k0 >> 5) & 1;
    if (wslot == 0 && (k0 + 32) < K)               // async-prefetch next slab
      tdm_load_a_tile(Aslab + k0 + 32, cur ? lds0 : lds1);

    const unsigned short* lA = &ldsA[cur][0];
    AB16 a[4], b[4];
#pragma unroll
    for (int t = 0; t < 4; ++t) {
      // A-operand from LDS slab (row-major 64x32):
      //   lane<16: K-local {0..7,16..23}, lane>=16: {8..15,24..31}
      const unsigned short* ap = lA + (t * 16 + lrow) * 32 + half * 8;
      a[t].q[0] = *(const uint4*)ap;
      a[t].q[1] = *(const uint4*)(ap + 16);
      // B-operand: lane holds one output column (a row of W), K-local {0..7,16..23} / {8..15,24..31}
      const unsigned short* bp = Wbase + (long long)(t * 16 + lrow) * K + k0 + half * 8;
      b[t].q[0] = *(const uint4*)bp;
      b[t].q[1] = *(const uint4*)(bp + 16);
    }
    if (k0 + 32 < K)
      __builtin_prefetch(Wbase + (long long)lrow * K + k0 + 32, 0, 0);
#pragma unroll
    for (int tm = 0; tm < 4; ++tm)
#pragma unroll
      for (int tn = 0; tn < 4; ++tn)
        acc[tm][tn] = __builtin_amdgcn_wmma_f32_16x16x32_bf16(
            false, a[tm].v, false, b[tn].v, (short)0, acc[tm][tn], false, false);

    if (wslot == 0)
      __builtin_amdgcn_s_wait_tensorcnt(0);        // next slab landed
    __syncthreads();                               // publish to all waves
  }

  unsigned short* obf = (unsigned short*)out;
  float*          of  = (float*)out;
#pragma unroll
  for (int tm = 0; tm < 4; ++tm) {
#pragma unroll
    for (int tn = 0; tn < 4; ++tn) {
      const int n  = n0 + tn * 16 + lrow;
      const float bv = bias[n];
#pragma unroll
      for (int r = 0; r < 8; ++r) {
        const int   m   = m0 + tm * 16 + r + half * 8;   // C layout: hi lanes = rows+8
        const float val = acc[tm][tn][r] + bv;
        if (mode == OUT_F32) {
          of[(long long)m * N + n] = val;
        } else {
          const int bb = m >> 11, s = m & (Sd - 1);
          const int h  = n >> 6,  d = n & 63;
          long long addr = (mode == OUT_QK)
              ? ((long long)(bb * Hd + h) * Sd + s) * HDd + d
              : ((long long)(bb * Hd + h) * HDd + d) * Sd + s;
          obf[addr] = f2bf(val);
        }
      }
    }
  }
}

// ---------------------------------------------------------------------------
// Flash attention: one wave per (b,h, 16-query tile). 32 keys per step.
//   Q,K: [B,H,S,HD] bf16;  Vt: [B,H,HD,S] bf16;  ctx out: [B*S, D] bf16
// ---------------------------------------------------------------------------
__global__ void __launch_bounds__(256)
flash_attn_wmma(const unsigned short* __restrict__ Q,
                const unsigned short* __restrict__ Kb,
                const unsigned short* __restrict__ Vt,
                unsigned short* __restrict__ ctx) {
  __shared__ unsigned short ldsP[8][16 * 32];   // per-wave 16x32 P tile (bf16)
  const int lane  = threadIdx.x & 31;
  const int half  = lane >> 4;
  const int lrow  = lane & 15;
  const int wslot = threadIdx.x >> 5;
  const int wg    = blockIdx.x * 8 + wslot;
  const int bh    = wg >> 7;                    // (b*H + h)
  const int q0    = (wg & 127) * 16;

  const unsigned short* Qh = Q  + (long long)bh * Sd * HDd;
  const unsigned short* Kh = Kb + (long long)bh * Sd * HDd;
  const unsigned short* Vh = Vt + (long long)bh * HDd * Sd;

  // Q A-operand tiles for both 32-wide K-steps over HD=64 (loaded once)
  AB16 aq[2];
#pragma unroll
  for (int ks = 0; ks < 2; ++ks) {
    const unsigned short* qp = Qh + (long long)(q0 + lrow) * HDd + ks * 32 + half * 8;
    aq[ks].q[0] = *(const uint4*)qp;
    aq[ks].q[1] = *(const uint4*)(qp + 16);
  }

  const v8f vzero = {0.f,0.f,0.f,0.f,0.f,0.f,0.f,0.f};
  v8f   accv[4];
  float mrow[8], lsum[8];
#pragma unroll
  for (int t = 0; t < 4; ++t) accv[t] = vzero;
#pragma unroll
  for (int r = 0; r < 8; ++r) { mrow[r] = -INFINITY; lsum[r] = 0.f; }

  const float scale = 0.125f;                   // 1/sqrt(64)

  for (int kb = 0; kb < Sd; kb += 32) {
    // scores tile: 16 queries x 32 keys (two 16-wide N tiles)
    v8f s0 = vzero, s1 = vzero;
#pragma unroll
    for (int ks = 0; ks < 2; ++ks) {
      AB16 b0, b1;
      const unsigned short* kp0 = Kh + (long long)(kb + lrow) * HDd + ks * 32 + half * 8;
      b0.q[0] = *(const uint4*)kp0;  b0.q[1] = *(const uint4*)(kp0 + 16);
      const unsigned short* kp1 = Kh + (long long)(kb + 16 + lrow) * HDd + ks * 32 + half * 8;
      b1.q[0] = *(const uint4*)kp1;  b1.q[1] = *(const uint4*)(kp1 + 16);
      s0 = __builtin_amdgcn_wmma_f32_16x16x32_bf16(false, aq[ks].v, false, b0.v, (short)0, s0, false, false);
      s1 = __builtin_amdgcn_wmma_f32_16x16x32_bf16(false, aq[ks].v, false, b1.v, (short)0, s1, false, false);
    }

    // online softmax over this 32-key block (rows live per C-layout halves)
    unsigned short* myP = ldsP[wslot];
    float alpha[8];
#pragma unroll
    for (int r = 0; r < 8; ++r) {
      const float a0 = s0[r] * scale, a1 = s1[r] * scale;
      float v = fmaxf(a0, a1);
      v = fmaxf(v, __shfl_xor(v, 1, 32));
      v = fmaxf(v, __shfl_xor(v, 2, 32));
      v = fmaxf(v, __shfl_xor(v, 4, 32));
      v = fmaxf(v, __shfl_xor(v, 8, 32));
      const float mn = fmaxf(mrow[r], v);
      const float al = __expf(mrow[r] - mn);
      const float p0 = __expf(a0 - mn);
      const float p1 = __expf(a1 - mn);
      float rs = p0 + p1;
      rs += __shfl_xor(rs, 1, 32);
      rs += __shfl_xor(rs, 2, 32);
      rs += __shfl_xor(rs, 4, 32);
      rs += __shfl_xor(rs, 8, 32);
      lsum[r] = lsum[r] * al + rs;
      mrow[r] = mn;
      alpha[r] = al;
      // write P (bf16) row-major 16x32 into this wave's LDS slab
      myP[(r + half * 8) * 32 + lrow]      = f2bf(p0);
      myP[(r + half * 8) * 32 + 16 + lrow] = f2bf(p1);
    }
#pragma unroll
    for (int td = 0; td < 4; ++td)
#pragma unroll
      for (int r = 0; r < 8; ++r) accv[td][r] *= alpha[r];

    // reload P in A-operand layout (k = key within block)
    AB16 pa;
    pa.q[0] = *(const uint4*)&myP[lrow * 32 + half * 8];
    pa.q[1] = *(const uint4*)&myP[lrow * 32 + 16 + half * 8];

    // ctx += P @ V  (4 N-tiles over HD=64; Vt gives contiguous key-dim per lane)
#pragma unroll
    for (int td = 0; td < 4; ++td) {
      AB16 vb;
      const unsigned short* vp = Vh + (long long)(td * 16 + lrow) * Sd + kb + half * 8;
      vb.q[0] = *(const uint4*)vp;  vb.q[1] = *(const uint4*)(vp + 16);
      accv[td] = __builtin_amdgcn_wmma_f32_16x16x32_bf16(
          false, pa.v, false, vb.v, (short)0, accv[td], false, false);
    }
  }

  // normalize and store ctx as bf16 [B*S, D] (head h occupies cols h*64..h*64+63)
  const int b = bh >> 4, h = bh & 15;
#pragma unroll
  for (int td = 0; td < 4; ++td) {
#pragma unroll
    for (int r = 0; r < 8; ++r) {
      const int s   = q0 + r + half * 8;
      const float v = accv[td][r] / lsum[r];
      ctx[((long long)(b * Sd + s)) * Dd + h * HDd + td * 16 + lrow] = f2bf(v);
    }
  }
}

// ---------------------------------------------------------------------------
// Launcher
// ---------------------------------------------------------------------------
extern "C" void kernel_launch(void* const* d_in, const int* in_sizes, int n_in,
                              void* d_out, int out_size, void* d_ws, size_t ws_size,
                              hipStream_t stream) {
  const float* x  = (const float*)d_in[0];
  const float* Wq = (const float*)d_in[1];
  const float* bq = (const float*)d_in[2];
  const float* Wk = (const float*)d_in[3];
  const float* bk = (const float*)d_in[4];
  const float* Wv = (const float*)d_in[5];
  const float* bv = (const float*)d_in[6];
  const float* Wp = (const float*)d_in[7];
  const float* bp = (const float*)d_in[8];
  const float* pe = (const float*)d_in[9];

  char* ws = (char*)d_ws;
  const size_t MB = 1024ull * 1024ull;
  unsigned short* Wq_bf = (unsigned short*)(ws + 0 * MB);   // 2 MB each
  unsigned short* Wk_bf = (unsigned short*)(ws + 2 * MB);
  unsigned short* Wv_bf = (unsigned short*)(ws + 4 * MB);
  unsigned short* Wp_bf = (unsigned short*)(ws + 6 * MB);
  unsigned short* xp    = (unsigned short*)(ws + 8 * MB);   // 16 MB
  unsigned short* Qb    = (unsigned short*)(ws + 24 * MB);  // 16 MB [B,H,S,HD]
  unsigned short* Kbuf  = (unsigned short*)(ws + 40 * MB);  // 16 MB [B,H,S,HD]
  unsigned short* Vtb   = (unsigned short*)(ws + 56 * MB);  // 16 MB [B,H,HD,S]
  unsigned short* ctxb  = (unsigned short*)(ws + 72 * MB);  // 16 MB [M,D]

  // weights fp32 -> bf16 (1M elems each, 4/thread)
  cvt_bf16<<<1024, 256, 0, stream>>>(Wq, Wq_bf);
  cvt_bf16<<<1024, 256, 0, stream>>>(Wk, Wk_bf);
  cvt_bf16<<<1024, 256, 0, stream>>>(Wv, Wv_bf);
  cvt_bf16<<<1024, 256, 0, stream>>>(Wp, Wp_bf);

  // xp = bf16(x + pe)  (8M elems, 4/thread)
  pe_add_bf16<<<8192, 256, 0, stream>>>(x, pe, xp);

  // projections: 2048 waves (64x64 tiles), 8 waves/block -> 256 blocks
  gemm_bf16_wmma<<<256, 256, 0, stream>>>(xp, Wq_bf, bq, (void*)Qb,  OUT_QK);
  gemm_bf16_wmma<<<256, 256, 0, stream>>>(xp, Wk_bf, bk, (void*)Kbuf, OUT_QK);
  gemm_bf16_wmma<<<256, 256, 0, stream>>>(xp, Wv_bf, bv, (void*)Vtb, OUT_VT);

  // attention: 64 (b,h) * 128 q-tiles = 8192 waves -> 1024 blocks
  flash_attn_wmma<<<1024, 256, 0, stream>>>(Qb, Kbuf, Vtb, ctxb);

  // output projection -> f32 d_out
  gemm_bf16_wmma<<<256, 256, 0, stream>>>(ctxb, Wp_bf, bp, d_out, OUT_F32);
}
